// DepTreeLSTM_12343736009154
// MI455X (gfx1250) — matrix operand order repaired and checked by reading, not compiled, not run on hardware
//
#include <hip/hip_runtime.h>
#include <hip/hip_bf16.h>
#include <math.h>

// ---------------------------------------------------------------------------
// TreeLSTM topological sweep on gfx1250 (wave32, WMMA f16->f32).
// Fused per-level GEMM: [M,512] x [512,640]
//   rows:  [ emb(256) | ht0(128) | ht1(128) ]
//   cols (packed, channel-interleaved): wave g owns channels [g*32, g*32+32)
//   of ALL five gate slices {i, o, u, g0, g1}, so the gate math needs no LDS.
// B fragments for one k-step are contiguous: loads use immediate offsets off
// one base pointer bumped once per k-step (no per-fragment address math).
// ---------------------------------------------------------------------------

#define T_TREES 4096
#define S_NODES 63
#define NN      (T_TREES * S_NODES)   // 258048
#define K_DIM   512
#define N_DIM   640
#define M0_MAX  131072                // nodes at level 0

typedef __attribute__((ext_vector_type(16))) _Float16 v16h;
typedef __attribute__((ext_vector_type(8)))  float    v8f;

union FragU { v16h v; uint4 q[2]; };

__device__ __forceinline__ float sigmoidf_(float x) {
    return 1.0f / (1.0f + __expf(-x));
}

// ---------------------------------------------------------------------------
// Pack weights into WMMA B-fragment order with channel-interleaved columns.
// Fragment (kt, fi=g*10+nt): 512 halves; lane l holds packed column
//   coll = l&15, halves j=0..15 hold K = kt*32 + (l>=16 ? 8 : 0) + (j<8 ? j : 8+j).
// Packed column (g, nt, coll) -> gate t = nt>>1, channel ch = g*32+(nt&1)*16+coll:
//   t<3 : column t*128+ch of [W_iou ; U_iou]
//   t==3: g0 = W_f[:,ch] ; U_f_w[:,ch]        (bias U_f_b[ch]     + b_f[ch])
//   t==4: g1 = W_f[:,ch] ; U_f_w[:,128+ch]    (bias U_f_b[128+ch] + b_f[ch])
// ---------------------------------------------------------------------------
__global__ void pack_weights(const float* __restrict__ W_iou, const float* __restrict__ U_iou,
                             const float* __restrict__ b_iou, const float* __restrict__ W_f,
                             const float* __restrict__ U_f_w, const float* __restrict__ U_f_b,
                             const float* __restrict__ b_f,
                             _Float16* __restrict__ wpack, float* __restrict__ biasp) {
    int tid = blockIdx.x * 256 + threadIdx.x;
    const int total = K_DIM * N_DIM; // 327680
    if (tid < total) {
        int j    = tid & 15;
        int lane = (tid >> 4) & 31;
        int frag = tid >> 9;          // kt*40 + fi
        int fi   = frag % 40;
        int kt   = frag / 40;
        int g    = fi / 10;
        int nt   = fi % 10;
        int k    = kt * 32 + ((lane >> 4) & 1) * 8 + (j < 8 ? j : 8 + j);
        int coll = lane & 15;
        int t    = nt >> 1;
        int ch   = g * 32 + (nt & 1) * 16 + coll;
        float v;
        if (t < 3) {
            int n = t * 128 + ch;
            v = (k < 256) ? W_iou[k * 384 + n] : U_iou[(k - 256) * 384 + n];
        } else {
            int uc = (t == 4) ? (128 + ch) : ch;
            v = (k < 256) ? W_f[k * 128 + ch] : U_f_w[(k - 256) * 256 + uc];
        }
        wpack[tid] = (_Float16)v;
    } else if (tid < total + N_DIM) {
        int pid  = tid - total;       // fi*16 + coll
        int fi   = pid >> 4;
        int coll = pid & 15;
        int g    = fi / 10;
        int nt   = fi % 10;
        int t    = nt >> 1;
        int ch   = g * 32 + (nt & 1) * 16 + coll;
        float b;
        if (t < 3)       b = b_iou[t * 128 + ch];
        else if (t == 3) b = U_f_b[ch] + b_f[ch];
        else             b = U_f_b[128 + ch] + b_f[ch];
        biasp[pid] = b;
    }
}

// ---------------------------------------------------------------------------
// Per-level gather: build f16 A rows [emb | ht0 | ht1]. One thread per (m,h).
// ---------------------------------------------------------------------------
__global__ void level_gather(const float* __restrict__ emb, const float* __restrict__ hbuf,
                             const int* __restrict__ children_idx, const int* __restrict__ child_type,
                             const float* __restrict__ child_mask,
                             _Float16* __restrict__ x2, int Md, int offs, int lgc) {
    int idx = blockIdx.x * 256 + threadIdx.x;
    if (idx >= Md * 128) return;
    int m = idx >> 7;
    int h = idx & 127;
    int tree = m >> lgc;
    int j    = m & ((1 << lgc) - 1);
    int node = tree * S_NODES + offs + j;

    float ht0 = 0.0f, ht1 = 0.0f;
#pragma unroll
    for (int c = 0; c < 2; ++c) {
        int   ch = children_idx[node * 2 + c];
        int   ty = child_type[node * 2 + c];
        float cm = child_mask[node * 2 + c];
        float hv = hbuf[(size_t)ch * 128 + h] * cm;   // masked: stale h * 0 at leaves
        if (ty == 0) ht0 += hv; else ht1 += hv;
    }
    _Float16* row = x2 + (size_t)m * K_DIM;
    row[2 * h]     = (_Float16)emb[(size_t)node * 256 + 2 * h];
    row[2 * h + 1] = (_Float16)emb[(size_t)node * 256 + 2 * h + 1];
    row[256 + h]   = (_Float16)ht0;
    row[384 + h]   = (_Float16)ht1;
}

// ---------------------------------------------------------------------------
// Per-level GEMM (WMMA) + fused gates. 128 threads = 4 waves, 32-row tile.
// Wave g: 2 M-subtiles x 10 n-tiles (160 acc VGPRs); every B fragment feeds
// two WMMAs. Epilogue is LDS-free: each wave owns all 5 gates for its 32
// channels, held in its own accumulators (WMMA C/D layout).
// ---------------------------------------------------------------------------
__global__ void __launch_bounds__(128)
level_gemm(const _Float16* __restrict__ x2, const _Float16* __restrict__ wpack,
           const float* __restrict__ biasp,
           const int* __restrict__ children_idx, const int* __restrict__ child_type,
           const float* __restrict__ child_mask,
           float* __restrict__ hbuf, float* __restrict__ cbuf, int offs, int lgc) {
    int tid  = threadIdx.x;
    int lane = tid & 31;
    int g    = tid >> 5;                 // 0..3 channel group
    int r0   = blockIdx.x * 32;          // first node-row of this tile
    int mrow = lane & 15;
    int khi  = (lane >> 4) & 1;
    int coll = lane & 15;
    int rowl = khi << 3;

    v8f acc[2][10] = {};

    // A: 16-bit A layout => two 16B K-runs per lane; advance 64B per k-step.
    const _Float16* aptr0 = x2 + (size_t)(r0 + mrow) * K_DIM + khi * 8;
    const _Float16* aptr1 = aptr0 + 16 * K_DIM;
    // B: 10 consecutive fragments per k-step; advance 40 fragments per k-step.
    const _Float16* bptr  = wpack + ((size_t)(g * 10) * 32 + lane) * 16;

#pragma unroll 1
    for (int kt = 0; kt < 16; ++kt) {
        FragU a0, a1;
        a0.q[0] = *(const uint4*)(aptr0);
        a0.q[1] = *(const uint4*)(aptr0 + 16);
        a1.q[0] = *(const uint4*)(aptr1);
        a1.q[1] = *(const uint4*)(aptr1 + 16);
#pragma unroll
        for (int nt = 0; nt < 10; ++nt) {
            FragU b;
            b.q[0] = *(const uint4*)(bptr + nt * 512);      // imm offset nt*1024B
            b.q[1] = *(const uint4*)(bptr + nt * 512 + 8);  // imm offset +16B
            acc[0][nt] = __builtin_amdgcn_wmma_f32_16x16x32_f16(
                false, a0.v, false, b.v, (short)0, acc[0][nt], false, false);
            acc[1][nt] = __builtin_amdgcn_wmma_f32_16x16x32_f16(
                false, a1.v, false, b.v, (short)0, acc[1][nt], false, false);
        }
        aptr0 += 32;          // 64 bytes: next 32-K slab
        aptr1 += 32;
        bptr  += 40 * 512;    // 40960 bytes: next k-step's fragment row
    }

    float bn[10];
#pragma unroll
    for (int nt = 0; nt < 10; ++nt)
        bn[nt] = biasp[(g * 10 + nt) * 16 + coll];

    // Fused gates, fully in registers (C/D layout: vgpr r, lanes>=16 = rows 8..15)
#pragma unroll
    for (int ms = 0; ms < 2; ++ms) {
#pragma unroll
        for (int r = 0; r < 8; ++r) {
            int row  = r0 + ms * 16 + rowl + r;
            int tree = row >> lgc;
            int jj   = row & ((1 << lgc) - 1);
            int node = tree * S_NODES + offs + jj;

            int   c0  = children_idx[node * 2];
            int   c1  = children_idx[node * 2 + 1];
            int   ty0 = child_type[node * 2];
            int   ty1 = child_type[node * 2 + 1];
            float cm0 = child_mask[node * 2];
            float cm1 = child_mask[node * 2 + 1];
#pragma unroll
            for (int p = 0; p < 2; ++p) {
                int ch = g * 32 + p * 16 + coll;
                float iv = acc[ms][0 + p][r] + bn[0 + p];
                float ov = acc[ms][2 + p][r] + bn[2 + p];
                float uv = acc[ms][4 + p][r] + bn[4 + p];
                float g0 = acc[ms][6 + p][r] + bn[6 + p];
                float g1 = acc[ms][8 + p][r] + bn[8 + p];
                float f0 = sigmoidf_(ty0 ? g1 : g0);
                float f1 = sigmoidf_(ty1 ? g1 : g0);
                float c_cell = f0 * cbuf[(size_t)c0 * 128 + ch] * cm0
                             + f1 * cbuf[(size_t)c1 * 128 + ch] * cm1;
                float c_new = sigmoidf_(iv) * tanhf(uv) + c_cell;
                float h_new = sigmoidf_(ov) * tanhf(c_new);
                cbuf[(size_t)node * 128 + ch] = c_new;
                hbuf[(size_t)node * 128 + ch] = h_new;
            }
        }
    }
}

// ---------------------------------------------------------------------------
extern "C" void kernel_launch(void* const* d_in, const int* in_sizes, int n_in,
                              void* d_out, int out_size, void* d_ws, size_t ws_size,
                              hipStream_t stream) {
    const float* emb          = (const float*)d_in[0];
    const float* child_mask   = (const float*)d_in[1];
    const float* W_iou        = (const float*)d_in[2];
    const float* U_iou        = (const float*)d_in[3];
    const float* b_iou        = (const float*)d_in[4];
    const float* W_f          = (const float*)d_in[5];
    const float* U_f_w        = (const float*)d_in[6];
    const float* U_f_b        = (const float*)d_in[7];
    const float* b_f          = (const float*)d_in[8];
    const int*   children_idx = (const int*)d_in[9];
    const int*   child_type   = (const int*)d_in[10];

    float* hbuf = (float*)d_out;                       // h state lives in the output

    char*  ws     = (char*)d_ws;
    size_t cbytes = (size_t)NN * 128 * sizeof(float);          // c state
    size_t xbytes = (size_t)M0_MAX * K_DIM * sizeof(_Float16); // A rows (max level)
    float*     cbuf  = (float*)ws;
    _Float16*  x2    = (_Float16*)(ws + cbytes);
    _Float16*  wpack = (_Float16*)(ws + cbytes + xbytes);
    float*     biasp = (float*)(ws + cbytes + xbytes + (size_t)K_DIM * N_DIM * sizeof(_Float16));

    {
        int total = K_DIM * N_DIM + N_DIM;
        pack_weights<<<(total + 255) / 256, 256, 0, stream>>>(
            W_iou, U_iou, b_iou, W_f, U_f_w, U_f_b, b_f, wpack, biasp);
    }

    static const int offs_tab[6] = {0, 32, 48, 56, 60, 62};
    for (int d = 0; d <= 5; ++d) {
        int lgc = 5 - d;
        int Md  = T_TREES << lgc;   // 131072 >> d
        level_gather<<<(Md * 128 + 255) / 256, 256, 0, stream>>>(
            emb, hbuf, children_idx, child_type, child_mask, x2, Md, offs_tab[d], lgc);
        level_gemm<<<Md / 32, 128, 0, stream>>>(
            x2, wpack, biasp, children_idx, child_type, child_mask,
            hbuf, cbuf, offs_tab[d], lgc);
    }
}
